// I2BGNN_50225347559911
// MI455X (gfx1250) — matrix operand order repaired and checked by reading, not compiled, not run on hardware
//
#include <hip/hip_runtime.h>
#include <hip/hip_bf16.h>
#include <math.h>

typedef _Float16 v16h __attribute__((ext_vector_type(16)));
typedef float    v8f  __attribute__((ext_vector_type(8)));

#define DIM 64

// ---------------------------------------------------------------------------
// WMMA GEMM: Y[Npad x 64] = X[N x KDIM] @ W[KDIM x 64], f32 in, f16 multiply,
// f32 accumulate via v_wmma_f32_16x16x32_f16.  Block = 128 threads (4 waves),
// block tile = 64 rows x 64 cols, wave tile = 16 rows x 64 cols (4 acc tiles).
// Y must be padded to a multiple of 64 rows (stores are unconditional).
// ---------------------------------------------------------------------------
template<int KDIM, int KPAD>
__global__ void __launch_bounds__(128)
gemm_wmma(const float* __restrict__ X, const float* __restrict__ W,
          float* __restrict__ Y, int nRows)
{
    __shared__ _Float16 sX [64 * KPAD];   // X block, row-major, zero-padded
    __shared__ _Float16 sWt[64 * KPAD];   // W transposed: sWt[n*KPAD + k]

    const int tid      = threadIdx.x;
    const int blockRow = blockIdx.x * 64;

    // Stage W (transposed) into LDS as f16, zero-padding K to KPAD.
    for (int idx = tid; idx < 64 * KPAD; idx += 128) {
        int n = idx / KPAD, k = idx % KPAD;
        sWt[idx] = (k < KDIM) ? (_Float16)W[k * 64 + n] : (_Float16)0.0f;
    }
    // Stage X block rows into LDS as f16 (pad OOB rows / K with zeros).
    for (int idx = tid; idx < 64 * KPAD; idx += 128) {
        int r = idx / KPAD, k = idx % KPAD;
        int grow = blockRow + r;
        float v = (grow < nRows && k < KDIM) ? X[(long)grow * KDIM + k] : 0.0f;
        sX[idx] = (_Float16)v;
    }
    __syncthreads();

    const int wave = tid >> 5;
    const int lane = tid & 31;
    const int lo   = lane & 15;   // row (A) / col (B,C)
    const int hi   = lane >> 4;   // half-select

    v8f acc[4] = {};

#pragma unroll
    for (int kc = 0; kc < KPAD / 32; ++kc) {
        const int k0 = kc * 32;
        // A: 16x32 f16; lane lo holds row M=lo; VGPR v packs K = kb, kb+1
        // with kb = (v>>2)*16 + hi*8 + (v&3)*2   (ISA 16-bit A 16x32 table)
        union { v16h v; unsigned u[8]; } A;
        const _Float16* xrow = &sX[(wave * 16 + lo) * KPAD];
#pragma unroll
        for (int v = 0; v < 8; ++v) {
            int kb = k0 + ((v >> 2) * 16) + hi * 8 + (v & 3) * 2;
            A.u[v] = *(const unsigned*)&xrow[kb];
        }
#pragma unroll
        for (int t = 0; t < 4; ++t) {
            // B: 32x16 f16; lane lo = col N; VGPR v packs K = (v>>2)*16 + hi*8 + (v&3)*2, +1
            union { v16h v; unsigned u[8]; } B;
            const _Float16* wcol = &sWt[(t * 16 + lo) * KPAD];
#pragma unroll
            for (int v = 0; v < 8; ++v) {
                int kb = k0 + ((v >> 2) * 16) + hi * 8 + (v & 3) * 2;
                B.u[v] = *(const unsigned*)&wcol[kb];
            }
            acc[t] = __builtin_amdgcn_wmma_f32_16x16x32_f16(
                /*neg_a=*/false, A.v, /*neg_b=*/false, B.v,
                /*c_mod=*/(short)0, acc[t],
                /*reuse_a=*/false, /*reuse_b=*/false);
        }
    }

    // C/D layout: lane lo = col, VGPR r -> row = hi*8 + r.
    // Y is padded to 64-row multiple: store unconditionally; all 32 stores
    // share one base address with immediate offsets (r*256 + t*64 bytes).
    float* ybase = &Y[(long)(blockRow + wave * 16 + hi * 8) * 64 + lo];
#pragma unroll
    for (int t = 0; t < 4; ++t) {
#pragma unroll
        for (int r = 0; r < 8; ++r) {
            ybase[r * 64 + t * 16] = acc[t][r];
        }
    }
}

// ---------------------------------------------------------------------------
// Elementwise / graph kernels
// ---------------------------------------------------------------------------
__global__ void fill_kernel(float* __restrict__ p, int n, float v)
{
    for (int i = blockIdx.x * blockDim.x + threadIdx.x; i < n;
         i += gridDim.x * blockDim.x) p[i] = v;
}

__global__ void deg_kernel(const int* __restrict__ ei, const float* __restrict__ ea,
                           float* __restrict__ deg, int E)
{
    for (int e = blockIdx.x * blockDim.x + threadIdx.x; e < E;
         e += gridDim.x * blockDim.x)
        atomicAdd(&deg[ei[E + e]], ea[2 * e + 1]);
}

__global__ void dinv_kernel(float* __restrict__ deg, int n)
{
    for (int i = blockIdx.x * blockDim.x + threadIdx.x; i < n;
         i += gridDim.x * blockDim.x) {
        float d = deg[i];
        deg[i] = (d > 0.0f) ? rsqrtf(d) : 0.0f;
    }
}

__global__ void norm_kernel(const int* __restrict__ ei, const float* __restrict__ ea,
                            const float* __restrict__ dinv, float* __restrict__ nrm, int E)
{
    for (int e = blockIdx.x * blockDim.x + threadIdx.x; e < E;
         e += gridDim.x * blockDim.x)
        nrm[e] = dinv[ei[e]] * ea[2 * e + 1] * dinv[ei[E + e]];
}

// agg[i][c] = hW[i][c] * dinv[i]^2   (self-loop term)
__global__ void init_agg(const float* __restrict__ hW, const float* __restrict__ dinv,
                         float* __restrict__ agg, int total)
{
    for (int idx = blockIdx.x * blockDim.x + threadIdx.x; idx < total;
         idx += gridDim.x * blockDim.x) {
        float di = dinv[idx >> 6];
        agg[idx] = hW[idx] * di * di;
    }
}

// One wave per edge: 32 lanes cover 64 channels as float2. Coalesced gather
// of hW[src] (L2-resident: 25.6 MB << 192 MB L2), coalesced f32 atomics.
__global__ void scatter_kernel(const float* __restrict__ hW, const int* __restrict__ ei,
                               const float* __restrict__ nrm, float* __restrict__ agg, int E)
{
    int gwave  = (blockIdx.x * blockDim.x + threadIdx.x) >> 5;
    int nwaves = (gridDim.x * blockDim.x) >> 5;
    int lane   = threadIdx.x & 31;
    for (int e = gwave; e < E; e += nwaves) {
        int   s = ei[e];
        int   d = ei[E + e];
        float w = nrm[e];
        if (e + nwaves < E)
            __builtin_prefetch(&hW[(long)ei[e + nwaves] * 64 + lane * 2], 0, 0);
        float2 v = *(const float2*)&hW[(long)s * 64 + lane * 2];
        float* dst = &agg[(long)d * 64 + lane * 2];
        atomicAdd(dst,     v.x * w);
        atomicAdd(dst + 1, v.y * w);
    }
}

// z = relu(agg + b) in place; accumulate per-channel sum / sumsq.
__global__ void __launch_bounds__(256)
bias_relu_stats(float* __restrict__ h, const float* __restrict__ bias,
                float* __restrict__ sums, float* __restrict__ sumsq, int total)
{
    __shared__ float s1[256], s2[256];
    int c = threadIdx.x & 63;       // stride is a multiple of 64 -> c fixed
    float b = bias[c];
    float lsum = 0.0f, lsq = 0.0f;
    for (int idx = blockIdx.x * 256 + threadIdx.x; idx < total;
         idx += gridDim.x * 256) {
        float z = h[idx] + b;
        z = z > 0.0f ? z : 0.0f;
        h[idx] = z;
        lsum += z; lsq += z * z;
    }
    s1[threadIdx.x] = lsum; s2[threadIdx.x] = lsq;
    __syncthreads();
    if (threadIdx.x < 64) {
        float a = s1[c] + s1[c + 64] + s1[c + 128] + s1[c + 192];
        float q = s2[c] + s2[c + 64] + s2[c + 128] + s2[c + 192];
        atomicAdd(&sums[c],  a);
        atomicAdd(&sumsq[c], q);
    }
}

__global__ void bn_apply(float* __restrict__ h, const float* __restrict__ sums,
                         const float* __restrict__ sumsq, const float* __restrict__ gamma,
                         const float* __restrict__ beta, int nNodes, int total)
{
    float invN = 1.0f / (float)nNodes;
    for (int idx = blockIdx.x * blockDim.x + threadIdx.x; idx < total;
         idx += gridDim.x * blockDim.x) {
        int c = idx & 63;
        float mu  = sums[c] * invN;
        float var = sumsq[c] * invN - mu * mu;
        float inv = rsqrtf(var + 1e-5f);
        h[idx] = gamma[c] * (h[idx] - mu) * inv + beta[c];
    }
}

__global__ void pool_kernel(const float* __restrict__ h, const int* __restrict__ batch,
                            float* __restrict__ pooled, int total)
{
    for (int idx = blockIdx.x * blockDim.x + threadIdx.x; idx < total;
         idx += gridDim.x * blockDim.x) {
        int i = idx >> 6, c = idx & 63;
        atomicAdd(&pooled[batch[i] * 64 + c], h[idx]);
    }
}

__global__ void __launch_bounds__(64)
head_kernel(const float* __restrict__ pooled, const float* __restrict__ w1,
            const float* __restrict__ b1, const float* __restrict__ w2,
            const float* __restrict__ b2, float* __restrict__ out)
{
    __shared__ float p[64], o1[64], o2[2];
    int g = blockIdx.x, c = threadIdx.x;
    p[c] = pooled[g * 64 + c];
    __syncthreads();
    float a = b1[c];
    for (int k = 0; k < 64; ++k) a += p[k] * w1[k * 64 + c];
    o1[c] = a > 0.0f ? a : 0.0f;
    __syncthreads();
    if (c < 2) {
        float s = b2[c];
        for (int k = 0; k < 64; ++k) s += o1[k] * w2[k * 2 + c];
        o2[c] = s;
    }
    __syncthreads();
    if (c < 2) {
        float m   = fmaxf(o2[0], o2[1]);
        float lse = m + logf(expf(o2[0] - m) + expf(o2[1] - m));
        out[g * 2 + c] = o2[c] - lse;
    }
}

// ---------------------------------------------------------------------------
static inline int cdiv(int a, int b) { return (a + b - 1) / b; }

extern "C" void kernel_launch(void* const* d_in, const int* in_sizes, int n_in,
                              void* d_out, int out_size, void* d_ws, size_t ws_size,
                              hipStream_t stream)
{
    (void)n_in; (void)ws_size;
    const float* x      = (const float*)d_in[0];
    const int*   ei     = (const int*)  d_in[1];
    const int*   batch  = (const int*)  d_in[2];
    const float* ea     = (const float*)d_in[3];
    const float* W0     = (const float*)d_in[4];
    const float* b0     = (const float*)d_in[5];
    const float* Ws     = (const float*)d_in[6];
    const float* bs     = (const float*)d_in[7];
    const float* gammas = (const float*)d_in[8];
    const float* betas  = (const float*)d_in[9];
    const float* l1w    = (const float*)d_in[10];
    const float* l1b    = (const float*)d_in[11];
    const float* l2w    = (const float*)d_in[12];
    const float* l2b    = (const float*)d_in[13];
    float* out = (float*)d_out;

    const int N = in_sizes[0] / 8;   // x: (N, 8)
    const int E = in_sizes[1] / 2;   // edge_index: (2, E)
    const int G = out_size / 2;      // out: (G, 2)
    const int total  = N * DIM;
    const int nBlk   = cdiv(N, 64);
    const int Npad   = nBlk * 64;    // GEMM stores unconditionally into pad

    float* ws     = (float*)d_ws;
    float* hA     = ws;                           // Npad*64 (h / agg / z)
    float* hB     = hA + (size_t)Npad * DIM;      // Npad*64 (h @ W)
    float* nrm    = hB + (size_t)Npad * DIM;      // E
    float* deg    = nrm + (size_t)E;              // N (deg, then dinv in place)
    float* sums   = deg + (size_t)N;              // 64
    float* sumsq  = sums + 64;                    // 64
    float* pooled = sumsq + 64;                   // G*64

    // --- symmetric normalization coefficients ---
    fill_kernel <<<cdiv(N, 256), 256, 0, stream>>>(deg, N, 1.0f);   // self-loop
    deg_kernel  <<<2048, 256, 0, stream>>>(ei, ea, deg, E);
    dinv_kernel <<<cdiv(N, 256), 256, 0, stream>>>(deg, N);
    norm_kernel <<<cdiv(E, 256), 256, 0, stream>>>(ei, ea, deg, nrm, E);

    // --- 3 GCN layers ---
    for (int layer = 0; layer < 3; ++layer) {
        const float* Wl = (layer == 0) ? W0 : Ws + (size_t)(layer - 1) * DIM * DIM;
        const float* bl = (layer == 0) ? b0 : bs + (size_t)(layer - 1) * DIM;

        if (layer == 0)
            gemm_wmma<8, 32>  <<<nBlk, 128, 0, stream>>>(x,  Wl, hB, N);
        else
            gemm_wmma<64, 64> <<<nBlk, 128, 0, stream>>>(hA, Wl, hB, N);

        init_agg       <<<cdiv(total, 256), 256, 0, stream>>>(hB, deg, hA, total);
        scatter_kernel <<<2048, 256, 0, stream>>>(hB, ei, nrm, hA, E);
        fill_kernel    <<<1, 128, 0, stream>>>(sums, 128, 0.0f);  // sums+sumsq
        bias_relu_stats<<<512, 256, 0, stream>>>(hA, bl, sums, sumsq, total);
        bn_apply       <<<cdiv(total, 256), 256, 0, stream>>>(
            hA, sums, sumsq, gammas + (size_t)layer * DIM, betas + (size_t)layer * DIM,
            N, total);
    }

    // --- pooling + MLP head + log_softmax ---
    fill_kernel <<<cdiv(G * DIM, 256), 256, 0, stream>>>(pooled, G * DIM, 0.0f);
    pool_kernel <<<cdiv(total, 256), 256, 0, stream>>>(hA, batch, pooled, total);
    head_kernel <<<G, 64, 0, stream>>>(pooled, l1w, l1b, l2w, l2b, out);
}
